// LiDARBackbone_75179107549306
// MI455X (gfx1250) — matrix operand order repaired and checked by reading, not compiled, not run on hardware
//
#include <hip/hip_runtime.h>

// ---------------------------------------------------------------------------
// Types for CDNA5 WMMA (wave32): D(16x16 f32) = A(16x32 bf16) x B(32x16 bf16) + C
// ---------------------------------------------------------------------------
typedef __attribute__((ext_vector_type(16))) __bf16 v16bf;
typedef __attribute__((ext_vector_type(8)))  float  v8f;

union ABf {
  v16bf v;
  uint4 q[2];
  unsigned long long d[4];
  unsigned short s[16];
};

static __device__ __forceinline__ unsigned short f2bf(float f) {
  unsigned u = __float_as_uint(f);
  u += 0x7FFFu + ((u >> 16) & 1u);           // round-to-nearest-even
  return (unsigned short)(u >> 16);
}
static __device__ __forceinline__ float bf2f(unsigned short h) {
  return __uint_as_float(((unsigned)h) << 16);
}

static __device__ __forceinline__ v8f wmma_bf16(const ABf& a, const ABf& b, v8f c) {
  return __builtin_amdgcn_wmma_f32_16x16x32_bf16(
      /*neg_a=*/false, a.v, /*neg_b=*/false, b.v,
      /*c_mod=*/(short)0, c, /*reuse_a=*/false, /*reuse_b=*/false);
}

static __device__ __forceinline__ float bn_relu(float x, float g, float b) {
  const float inv_sqrt_1pe = 0.9999950000374996f;  // 1/sqrt(1+1e-5)
  float v = x * (g * inv_sqrt_1pe) + b;
  return v > 0.f ? v : 0.f;
}

// ---------------------------------------------------------------------------
// 1) Voxelize: scatter-add 2M points into channel-last fp32 grid [z][y][x][4]
// ---------------------------------------------------------------------------
__global__ void k_voxelize(const float* __restrict__ pts, float* __restrict__ grid, int N) {
  int i = blockIdx.x * blockDim.x + threadIdx.x;
  if (i >= N) return;
  float4 p = ((const float4*)pts)[i];
  bool ok = p.x >= -32.f && p.x <= 32.f &&
            p.y >= -32.f && p.y <= 32.f &&
            p.z >= -32.f && p.z <= 32.f;
  if (!ok) return;  // masked points contribute 0
  int cx = (int)floorf((p.x + 32.f) * 2.f); cx = min(max(cx, 0), 127);
  int cy = (int)floorf((p.y + 32.f) * 2.f); cy = min(max(cy, 0), 127);
  int cz = (int)floorf((p.z + 32.f) * 2.f); cz = min(max(cz, 0), 127);
  float* g = grid + ((((cz * 128 + cy) * 128) + cx) << 2);
  atomicAdd(g + 0, p.x);
  atomicAdd(g + 1, p.y);
  atomicAdd(g + 2, p.z);
  atomicAdd(g + 3, p.w);
}

// fp32 -> bf16 convert
__global__ void k_cvt(const float* __restrict__ in, unsigned short* __restrict__ out, int n) {
  int i = blockIdx.x * blockDim.x + threadIdx.x;
  if (i < n) out[i] = f2bf(in[i]);
}

// ---------------------------------------------------------------------------
// Weight prep
// Layer 1: wa1[co][k], k = tap*4 + ci, padded to K=128 with zeros (tap>=27)
// Layers 2-6: wbf[tap][co][ci] (bf16) from w[co][ci][3][3][3] (fp32)
// ---------------------------------------------------------------------------
__global__ void k_prep_w1(const float* __restrict__ w, unsigned short* __restrict__ wa) {
  int i = blockIdx.x * blockDim.x + threadIdx.x;  // 32*128
  if (i >= 32 * 128) return;
  int co = i >> 7, k = i & 127;
  int tap = k >> 2, ci = k & 3;
  float v = (tap < 27) ? w[(co * 4 + ci) * 27 + tap] : 0.f;
  wa[i] = f2bf(v);
}

__global__ void k_prep_w(const float* __restrict__ w, unsigned short* __restrict__ wbf,
                         int Co, int Ci) {
  int i = blockIdx.x * blockDim.x + threadIdx.x;
  int total = 27 * Co * Ci;
  if (i >= total) return;
  int ci = i % Ci;
  int r  = i / Ci;
  int co = r % Co;
  int t  = r / Co;
  wbf[i] = f2bf(w[(co * Ci + ci) * 27 + t]);
}

// ---------------------------------------------------------------------------
// Layer 1: Cin=4, Cout=32, D=128, s=1. Implicit GEMM, K padded 108->128.
// Block = 64 threads (2 waves: co tiles 0-15, 16-31). Grid: (8, 1, 128*128).
// ---------------------------------------------------------------------------
__global__ __launch_bounds__(64) void k_conv1(
    const unsigned short* __restrict__ act,   // [128^3][4] bf16
    const unsigned short* __restrict__ wa,    // [32][128]  bf16
    const float* __restrict__ gg, const float* __restrict__ bb,
    unsigned short* __restrict__ out) {       // [128^3][32] bf16
  const int D = 128;
  int lane = threadIdx.x & 31;
  int wave = threadIdx.x >> 5;
  int mn   = lane & 15;     // m for A/D, n for B/D
  int half = lane >> 4;
  int co0  = wave * 16;
  int x0   = blockIdx.x * 16;
  int zy   = blockIdx.z;
  int zo = zy / D, yo = zy % D;
  int xo = x0 + mn;

  v8f acc = {};
#pragma unroll
  for (int c0 = 0; c0 < 128; c0 += 32) {
    ABf a, b;
    const unsigned short* wp = wa + (co0 + mn) * 128 + c0 + half * 16;
    a.q[0] = *(const uint4*)(wp);
    a.q[1] = *(const uint4*)(wp + 8);
    int kb = c0 + half * 16;
#pragma unroll
    for (int tt = 0; tt < 4; ++tt) {
      int tap = (kb >> 2) + tt;
      unsigned long long v = 0ull;
      if (tap < 27) {
        int dz = tap / 9, r = tap % 9, dy = r / 3, dx = r % 3;
        int zi = zo + dz - 1, yi = yo + dy - 1, xi = xo + dx - 1;
        if ((unsigned)zi < 128u && (unsigned)yi < 128u && (unsigned)xi < 128u)
          v = *(const unsigned long long*)(act + ((((zi * 128 + yi) * 128) + xi) << 2));
      }
      b.d[tt] = v;
    }
    acc = wmma_bf16(a, b, acc);
  }

  int pos = (zo * 128 + yo) * 128 + xo;
  union { uint4 q; unsigned short s[8]; } o;
#pragma unroll
  for (int j = 0; j < 8; ++j) {
    int c = co0 + half * 8 + j;
    o.s[j] = f2bf(bn_relu(acc[j], gg[c], bb[c]));
  }
  *(uint4*)(out + pos * 32 + co0 + half * 8) = o.q;
}

// ---------------------------------------------------------------------------
// Layers 2-6: implicit-GEMM conv3d(k=3,p=1) + BN + ReLU, bf16 WMMA.
// 2x2 register blocking: each wave computes 32 Cout x 32 x-positions
// (four 16x16 WMMA tiles). Block = 128 threads = 4 waves arranged WM x WN over
// (Cout, x). Per (dz,dy) the block stages the ((WN*32-1)*S+3)-wide input strip
// directly into padded LDS with GLOBAL_LOAD_ASYNC_TO_LDS_B128 (ASYNCcnt, no
// VGPR round-trip); x-OOB positions are zero-filled with ds stores. B
// fragments are branchless 16B ds loads shared across waves and the 3 dx taps.
// Grid: x = DOUT/(WN*32), y = COUT/(WM*32), z = DOUT*DOUT (zo,yo).
// ---------------------------------------------------------------------------
template <int CIN, int COUT, int DIN, int DOUT, int S, int WM, int WN>
__global__ __launch_bounds__(128) void k_conv(
    const unsigned short* __restrict__ act,   // [DIN^3][CIN]  bf16
    const unsigned short* __restrict__ wbf,   // [27][COUT][CIN] bf16
    const float* __restrict__ gg, const float* __restrict__ bb,
    unsigned short* __restrict__ out) {       // [DOUT^3][COUT] bf16
  constexpr int NX = WN * 32;           // block x coverage
  constexpr int CP = CIN + 8;           // padded LDS row stride (elements)
  constexpr int W  = (NX - 1) * S + 3;  // staged input strip width
  constexpr int CH = CIN / 8;           // 16B chunks per position
  __shared__ unsigned short patch[W * CP];

  int tid  = threadIdx.x;
  int lane = tid & 31;
  int wave = tid >> 5;
  int wm   = wave % WM;                 // wave's Cout slot
  int wx   = wave / WM;                 // wave's x slot
  int mn   = lane & 15;
  int half = lane >> 4;
  int co0  = blockIdx.y * (WM * 32) + wm * 32;
  int xb   = blockIdx.x * NX;           // block x origin
  int zy   = blockIdx.z;
  int zo = zy / DOUT, yo = zy % DOUT;
  int xin0 = xb * S - 1;

  v8f acc00 = {}, acc01 = {}, acc10 = {}, acc11 = {};
#pragma unroll
  for (int dz = 0; dz < 3; ++dz) {
    int zi = zo * S + dz - 1;                       // block-uniform
    if ((unsigned)zi >= (unsigned)DIN) continue;    // uniform: EXEC stays full
#pragma unroll
    for (int dy = 0; dy < 3; ++dy) {
      int yi = yo * S + dy - 1;                     // block-uniform
      if ((unsigned)yi >= (unsigned)DIN) continue;  // uniform
      const unsigned short* row = act + (size_t)(zi * DIN + yi) * DIN * CIN;

      __syncthreads();  // protect LDS from previous iteration's readers
      for (int q = tid; q < W * CH; q += 128) {
        int n  = q / CH;
        int c8 = q - n * CH;
        int xi = xin0 + n;
        unsigned short* dst = patch + n * CP + c8 * 8;
        if ((unsigned)xi < (unsigned)DIN) {
          // Async DMA: memory -> LDS, 16B, tracked by ASYNCcnt.
          unsigned lds_off = (unsigned)(unsigned long long)(void*)dst;
          unsigned long long ga =
              (unsigned long long)(const void*)(row + (size_t)xi * CIN + c8 * 8);
          asm volatile("global_load_async_to_lds_b128 %0, %1, off"
                       :: "v"(lds_off), "v"(ga) : "memory");
        } else {
          uint4 z = {0, 0, 0, 0};
          *(uint4*)dst = z;
        }
      }
      // Prefetch next (dz,dy) row while this strip is consumed.
      if (dy < 2) {
        int yn = yo * S + dy;                       // next yi
        if ((unsigned)yn < (unsigned)DIN) {
          const char* pf = (const char*)(act + (size_t)(zi * DIN + yn) * DIN * CIN) + tid * 64;
          __builtin_prefetch(pf, 0, 1);
        }
      }
      asm volatile("s_wait_asynccnt 0x0" ::: "memory");
      __syncthreads();

      int tapzy = (dz * 3 + dy) * 3;
#pragma unroll
      for (int dx = 0; dx < 3; ++dx) {
        const unsigned short* wp =
            wbf + (size_t)((tapzy + dx) * COUT + co0 + mn) * CIN + half * 8;
        const unsigned short* bp =
            patch + ((wx * 32 + mn) * S + dx) * CP + half * 8;
#pragma unroll
        for (int c0 = 0; c0 < CIN; c0 += 32) {
          ABf a0, a1, b0, b1;
          a0.q[0] = *(const uint4*)(wp + c0);
          a0.q[1] = *(const uint4*)(wp + c0 + 16);
          a1.q[0] = *(const uint4*)(wp + 16 * CIN + c0);
          a1.q[1] = *(const uint4*)(wp + 16 * CIN + c0 + 16);
          b0.q[0] = *(const uint4*)(bp + c0);
          b0.q[1] = *(const uint4*)(bp + c0 + 16);
          b1.q[0] = *(const uint4*)(bp + 16 * S * CP + c0);
          b1.q[1] = *(const uint4*)(bp + 16 * S * CP + c0 + 16);
          acc00 = wmma_bf16(a0, b0, acc00);
          acc01 = wmma_bf16(a0, b1, acc01);
          acc10 = wmma_bf16(a1, b0, acc10);
          acc11 = wmma_bf16(a1, b1, acc11);
        }
      }
    }
  }

  // Epilogue: BN+ReLU, bf16 store; one 16B store per lane per tile.
  int posz = (zo * DOUT + yo) * DOUT + xb + wx * 32 + mn;
  v8f accs[2][2] = {{acc00, acc01}, {acc10, acc11}};
#pragma unroll
  for (int mi = 0; mi < 2; ++mi) {
#pragma unroll
    for (int ni = 0; ni < 2; ++ni) {
      int pos = posz + ni * 16;
      union { uint4 q; unsigned short s[8]; } o;
#pragma unroll
      for (int j = 0; j < 8; ++j) {
        int c = co0 + mi * 16 + half * 8 + j;
        o.s[j] = f2bf(bn_relu(accs[mi][ni][j], gg[c], bb[c]));
      }
      *(uint4*)(out + (size_t)pos * COUT + co0 + mi * 16 + half * 8) = o.q;
    }
  }
}

// ---------------------------------------------------------------------------
// Max over y: feat [z][y][x][256] (D=32) -> bev [z][x][256] fp32
// ---------------------------------------------------------------------------
__global__ void k_maxy(const unsigned short* __restrict__ act, float* __restrict__ bev) {
  int i = blockIdx.x * blockDim.x + threadIdx.x;   // (z*32+x)*256 + c
  if (i >= 32 * 32 * 256) return;
  int c = i & 255;
  int zx = i >> 8;
  int x = zx & 31, z = zx >> 5;
  float m = -3.4e38f;
#pragma unroll 4
  for (int y = 0; y < 32; ++y)
    m = fmaxf(m, bf2f(act[((((z * 32 + y) * 32) + x) << 8) + c]));
  bev[i] = m;
}

// ---------------------------------------------------------------------------
// Half-pixel bilinear resize (32,32) -> (200,200); out [c][i][j] fp32
// ---------------------------------------------------------------------------
__global__ void k_resize(const float* __restrict__ bev, float* __restrict__ out) {
  int i = blockIdx.x * blockDim.x + threadIdx.x;   // (c*200+ii)*200 + j
  if (i >= 256 * 200 * 200) return;
  int j = i % 200;
  int r = i / 200;
  int ii = r % 200;
  int c = r / 200;
  float sz = fminf(fmaxf((ii + 0.5f) * 0.16f - 0.5f, 0.f), 31.f);
  float sx = fminf(fmaxf((j  + 0.5f) * 0.16f - 0.5f, 0.f), 31.f);
  int z0 = (int)sz, x0 = (int)sx;
  int z1 = min(z0 + 1, 31), x1 = min(x0 + 1, 31);
  float fz = sz - (float)z0, fx = sx - (float)x0;
  float v00 = bev[((z0 * 32 + x0) << 8) + c];
  float v01 = bev[((z0 * 32 + x1) << 8) + c];
  float v10 = bev[((z1 * 32 + x0) << 8) + c];
  float v11 = bev[((z1 * 32 + x1) << 8) + c];
  float v0 = v00 + (v01 - v00) * fx;
  float v1 = v10 + (v11 - v10) * fx;
  out[i] = v0 + (v1 - v0) * fz;
}

// ---------------------------------------------------------------------------
// Launch
// Inputs (setup_inputs order): points, then (w1,g1,b1), (w2,g2,b2), ... (w6,g6,b6)
// ---------------------------------------------------------------------------
extern "C" void kernel_launch(void* const* d_in, const int* in_sizes, int n_in,
                              void* d_out, int out_size, void* d_ws, size_t ws_size,
                              hipStream_t stream) {
  (void)n_in; (void)out_size; (void)ws_size;
  const float* pts = (const float*)d_in[0];
  const float *w[6], *g[6], *b[6];
  for (int i = 0; i < 6; ++i) {
    w[i] = (const float*)d_in[1 + 3 * i];
    g[i] = (const float*)d_in[2 + 3 * i];
    b[i] = (const float*)d_in[3 + 3 * i];
  }

  // Workspace layout (bytes):
  //   [0, 32MB)       grid fp32 (4 * 128^3)
  //   [32MB, 64MB)    bufA bf16 (max 33.5MB: act0/act2/act4/act6)
  //   [64MB, 192MB)   bufB bf16 (max 134MB: act1/act3/act5)
  //   [192MB+, ...]   bf16 weights + fp32 bev
  char* ws = (char*)d_ws;
  float*          grid = (float*)ws;
  unsigned short* bufA = (unsigned short*)(ws + 33554432);
  unsigned short* bufB = (unsigned short*)(ws + 67108864);
  unsigned short* wa1  = (unsigned short*)(ws + 201326592);
  unsigned short* wb2  = wa1 + 32 * 128;
  unsigned short* wb3  = wb2 + 27 * 64 * 32;
  unsigned short* wb4  = wb3 + 27 * 64 * 64;
  unsigned short* wb5  = wb4 + 27 * 128 * 64;
  unsigned short* wb6  = wb5 + 27 * 128 * 128;
  float*          bev  = (float*)(ws + 201326592 + 4194304);

  int N = in_sizes[0] / 4;

  // Voxelize (grid must be re-zeroed every call; atomics accumulate)
  hipMemsetAsync(grid, 0, 33554432, stream);
  k_voxelize<<<(N + 255) / 256, 256, 0, stream>>>(pts, grid, N);
  k_cvt<<<(8388608 + 255) / 256, 256, 0, stream>>>(grid, bufA, 8388608);

  // Weight prep (bf16, WMMA-friendly layouts)
  k_prep_w1<<<(32 * 128 + 255) / 256, 256, 0, stream>>>(w[0], wa1);
  k_prep_w<<<(27 * 64 * 32 + 255) / 256, 256, 0, stream>>>(w[1], wb2, 64, 32);
  k_prep_w<<<(27 * 64 * 64 + 255) / 256, 256, 0, stream>>>(w[2], wb3, 64, 64);
  k_prep_w<<<(27 * 128 * 64 + 255) / 256, 256, 0, stream>>>(w[3], wb4, 128, 64);
  k_prep_w<<<(27 * 128 * 128 + 255) / 256, 256, 0, stream>>>(w[4], wb5, 128, 128);
  k_prep_w<<<(27 * 256 * 128 + 255) / 256, 256, 0, stream>>>(w[5], wb6, 256, 128);

  // Conv stack (bf16 WMMA implicit GEMM, BN+ReLU fused)
  k_conv1<<<dim3(8, 1, 128 * 128), 64, 0, stream>>>(bufA, wa1, g[0], b[0], bufB);
  k_conv<32, 64, 128, 64, 2, 2, 2>
      <<<dim3(1, 1, 64 * 64), 128, 0, stream>>>(bufB, wb2, g[1], b[1], bufA);
  k_conv<64, 64, 64, 64, 1, 2, 2>
      <<<dim3(1, 1, 64 * 64), 128, 0, stream>>>(bufA, wb3, g[2], b[2], bufB);
  k_conv<64, 128, 64, 32, 2, 4, 1>
      <<<dim3(1, 1, 32 * 32), 128, 0, stream>>>(bufB, wb4, g[3], b[3], bufA);
  k_conv<128, 128, 32, 32, 1, 4, 1>
      <<<dim3(1, 1, 32 * 32), 128, 0, stream>>>(bufA, wb5, g[4], b[4], bufB);
  k_conv<128, 256, 32, 32, 1, 4, 1>
      <<<dim3(1, 2, 32 * 32), 128, 0, stream>>>(bufB, wb6, g[5], b[5], bufA);

  // BEV max over y + bilinear upsample to 200x200
  k_maxy<<<(32 * 32 * 256) / 256, 256, 0, stream>>>(bufA, bev);
  k_resize<<<(256 * 200 * 200 + 255) / 256, 256, 0, stream>>>(bev, (float*)d_out);
}
